// MLA_52338471469334
// MI455X (gfx1250) — matrix operand order repaired and checked by reading, not compiled, not run on hardware
//
#include <hip/hip_runtime.h>

#define B_    2
#define S_    2048
#define DIM_  2048
#define DOWN_ 512
#define UP_   2048
#define H_    16
#define RD_   64
#define VHD_  128
#define QKD_  192

typedef __attribute__((ext_vector_type(16))) __bf16 v16bf;
typedef __attribute__((ext_vector_type(8)))  float  v8f;
typedef __attribute__((ext_vector_type(4)))  unsigned int v4u;
typedef __attribute__((ext_vector_type(8)))  unsigned int v8u;

__device__ __forceinline__ unsigned short f32_to_bf16(float f) {
  unsigned int u = __float_as_uint(f);
  u += 0x7fffu + ((u >> 16) & 1u);
  return (unsigned short)(u >> 16);
}

union Frag16 { v16bf v; uint4 q[2]; unsigned short s[16]; };

// LDS byte offset (wave-relative) of a __shared__ object: generic -> as(3) -> int
__device__ __forceinline__ unsigned lds_off_u32(const void* p) {
  return (unsigned)(unsigned long long)(__attribute__((address_space(3))) const void*)p;
}

// ---------------------------------------------------------------------------
// TDM: 2D tile load Global->LDS via tensor_load_to_lds (D# per ISA 08 §8).
// tile_d0 elems/row (bf16), tile_d1 rows, row stride = stride_elems.
// Issue from ONE wave (uniform operands -> SGPR groups), wait TENSORcnt.
// ---------------------------------------------------------------------------
__device__ __forceinline__ void tdm_load_2d_bf16(unsigned lds_off,
                                                 const unsigned short* gptr,
                                                 unsigned tile_d0, unsigned tile_d1,
                                                 unsigned tensor_d0, unsigned tensor_d1,
                                                 unsigned stride_elems) {
  unsigned long long ga = (unsigned long long)gptr;
  v4u g0;
  g0[0] = 1u;                                         // count=1, no gather
  g0[1] = lds_off;                                    // lds_addr (bytes)
  g0[2] = (unsigned)(ga & 0xffffffffu);               // global_addr[31:0]
  g0[3] = (unsigned)((ga >> 32) & 0x01ffffffu)        // global_addr[56:32]
          | (2u << 30);                               // type=2 ("image")
  v8u g1;
  g1[0] = (1u << 16);                                 // wg_mask=0, data_size=1 (2B)
  g1[1] = (tensor_d0 & 0xffffu) << 16;                // tensor_dim0[15:0]
  g1[2] = (tensor_d0 >> 16) | ((tensor_d1 & 0xffffu) << 16);
  g1[3] = (tensor_d1 >> 16) | (tile_d0 << 16);        // tile_dim0
  g1[4] = tile_d1 & 0xffffu;                          // tile_dim1, tile_dim2=0
  g1[5] = stride_elems;                               // tensor_dim0_stride[31:0]
  g1[6] = 0u;
  g1[7] = 0u;
  asm volatile("tensor_load_to_lds %0, %1" :: "s"(g0), "s"(g1) : "memory");
}

// Per-lane async copy: 16B Global -> LDS (ASYNCcnt-tracked)
__device__ __forceinline__ void async_b128(unsigned lds_byte, unsigned goff_byte,
                                           const void* base) {
  asm volatile("global_load_async_to_lds_b128 %0, %1, %2"
               :: "v"(lds_byte), "v"(goff_byte), "s"(base) : "memory");
}
__device__ __forceinline__ void wait_async0() {
  asm volatile("s_wait_asynccnt 0x0" ::: "memory");
}
__device__ __forceinline__ void wait_tensor0() {
  asm volatile("s_wait_tensorcnt 0x0" ::: "memory");
}

// ---------------------------------------------------------------------------
// f32 -> bf16 pack
// ---------------------------------------------------------------------------
__global__ void cvt_f32_bf16_kernel(const float* __restrict__ in,
                                    unsigned short* __restrict__ out, int n) {
  int i = blockIdx.x * 256 + threadIdx.x;
  if (i < n) out[i] = f32_to_bf16(in[i]);
}

// f32 [K,N] -> bf16 transposed [N,K] (one-time; lets GEMM B tiles async-copy)
__global__ void cvt_transpose_bf16_kernel(const float* __restrict__ in,
                                          unsigned short* __restrict__ out,
                                          int K, int N) {
  long long i = (long long)blockIdx.x * 256 + threadIdx.x;
  if (i < (long long)K * N) {
    int n = (int)(i / K);
    int k = (int)(i % K);
    out[i] = f32_to_bf16(in[(long long)k * N + n]);
  }
}

// ---------------------------------------------------------------------------
// Tiled bf16 GEMM:  C[M,N] = A[M,K] * W[K,N] + bias   (W given as WT[N,K])
// block = 256 threads (8 waves), tile 128x64, K-step 32.
// A tile staged by TDM (wave 0), WT tile by per-lane async b128.
// ---------------------------------------------------------------------------
#define GTM 128
#define GTN 64
#define GTK 32

__global__ __launch_bounds__(256) void gemm_bf16_kernel(
    const unsigned short* __restrict__ A,    // [M,K] bf16
    const unsigned short* __restrict__ WT,   // [N,K] bf16 (pre-transposed W)
    const float* __restrict__ bias,          // [N]
    float* __restrict__ Cf,                  // [M,N] f32 (may be null)
    unsigned short* __restrict__ Cbf,        // [M,N] bf16 (may be null)
    int M, int N, int K) {
  __shared__ __align__(16) unsigned short sA[GTM][GTK];    // 8 KB
  __shared__ __align__(16) unsigned short sBt[GTN][GTK];   // 4 KB

  const int tid   = threadIdx.x;
  const int wave  = tid >> 5;
  const int lane  = tid & 31;
  const int lhalf = lane >> 4;
  const int lmod  = lane & 15;
  const int bm = blockIdx.x * GTM;
  const int bn = blockIdx.y * GTN;

  const unsigned sA_off  = lds_off_u32(&sA[0][0]);
  const unsigned sBt_off = lds_off_u32(&sBt[0][0]);

  v8f acc[4] = {};

  for (int k0 = 0; k0 < K; k0 += GTK) {
    // --- A tile 128x32 via Tensor Data Mover (wave 0 only, SGPR descriptor) ---
    if (wave == 0) {
      tdm_load_2d_bf16(sA_off, A + (size_t)bm * K + k0,
                       /*tile*/ GTK, GTM,
                       /*tensor*/ (unsigned)(K - k0), (unsigned)(M - bm),
                       /*stride*/ (unsigned)K);
    }
    // --- WT tile 64x32 via async copy: thread t -> 16B chunk t ---
    {
      unsigned ldsb = sBt_off + (unsigned)(tid * 16);
      unsigned goff = (unsigned)((((size_t)(bn + (tid >> 2)) * K + k0 + (tid & 3) * 8) * 2));
      async_b128(ldsb, goff, WT);
    }
    wait_async0();
    if (wave == 0) wait_tensor0();
    __syncthreads();

    // --- A fragment (contiguous k order: k = 16*half + s) ---
    Frag16 ua;
    {
      int row = (wave << 4) + lmod;
      int kb  = lhalf << 4;
      const unsigned short* ap = &sA[row][0];
      ua.q[0] = *(const uint4*)(ap + kb);
      ua.q[1] = *(const uint4*)(ap + kb + 8);
    }
#pragma unroll
    for (int ns = 0; ns < 4; ++ns) {
      Frag16 ub;
      int n   = (ns << 4) + lmod;
      int kb2 = lhalf << 4;
      ub.q[0] = *(const uint4*)&sBt[n][kb2];
      ub.q[1] = *(const uint4*)&sBt[n][kb2 + 8];
      acc[ns] = __builtin_amdgcn_wmma_f32_16x16x32_bf16(
          false, ua.v, false, ub.v, (short)0, acc[ns], false, false);
    }
    __syncthreads();
  }

  int rowbase = bm + (wave << 4) + (lhalf << 3);
#pragma unroll
  for (int ns = 0; ns < 4; ++ns) {
    int col = bn + (ns << 4) + lmod;
    float bv = bias ? bias[col] : 0.0f;
#pragma unroll
    for (int e = 0; e < 8; ++e) {
      int row = rowbase + e;
      float v = acc[ns][e] + bv;
      if (Cf)  Cf[(size_t)row * N + col]  = v;
      if (Cbf) Cbf[(size_t)row * N + col] = f32_to_bf16(v);
    }
  }
}

// ---------------------------------------------------------------------------
// RoPE + concat + layout change to [B,H,S,D] bf16
// ---------------------------------------------------------------------------
__global__ __launch_bounds__(256) void assemble_qkv_kernel(
    const float* __restrict__ qc, const float* __restrict__ qpe,
    const float* __restrict__ kc, const float* __restrict__ kpe,
    const float* __restrict__ vf,
    const float* __restrict__ pcos, const float* __restrict__ psin,
    unsigned short* __restrict__ qb, unsigned short* __restrict__ kb,
    unsigned short* __restrict__ vb) {
  int bs = blockIdx.x;
  int b  = bs / S_;
  int s  = bs % S_;
  for (int idx = threadIdx.x; idx < H_ * QKD_; idx += 256) {
    int h = idx / QKD_;
    int d = idx % QKD_;
    size_t dst = ((size_t)(b * H_ + h) * S_ + s) * QKD_ + d;
    float qv, kv;
    if (d < VHD_) {
      qv = qc[(size_t)bs * UP_ + h * VHD_ + d];
      kv = kc[(size_t)bs * UP_ + h * VHD_ + d];
    } else {
      int dd = d - VHD_;
      int fi = dd >> 1;
      float c  = pcos[s * (RD_ / 2) + fi];
      float sn = psin[s * (RD_ / 2) + fi];
      const float* qp = qpe + (size_t)bs * (RD_ * H_) + h * RD_;
      const float* kp = kpe + (size_t)bs * RD_;
      if ((dd & 1) == 0) {
        qv = qp[dd] * c - qp[dd + 1] * sn;
        kv = kp[dd] * c - kp[dd + 1] * sn;
      } else {
        qv = qp[dd - 1] * sn + qp[dd] * c;
        kv = kp[dd - 1] * sn + kp[dd] * c;
      }
    }
    qb[dst] = f32_to_bf16(qv);
    kb[dst] = f32_to_bf16(kv);
  }
  for (int idx = threadIdx.x; idx < H_ * VHD_; idx += 256) {
    int h = idx >> 7;
    int d = idx & 127;
    vb[((size_t)(b * H_ + h) * S_ + s) * VHD_ + d] =
        f32_to_bf16(vf[(size_t)bs * UP_ + idx]);
  }
}

// ---------------------------------------------------------------------------
// Flash attention (causal, online softmax), 4 waves / 64 q-rows per block.
// K chunk (12KB contiguous) staged by async b128; V chunk transposed via DS;
// P converted through per-wave LDS; 12 score + 8 P*V WMMAs per 32-col step.
// ---------------------------------------------------------------------------
__global__ __launch_bounds__(128) void mla_attention_kernel(
    const unsigned short* __restrict__ qbf,   // [B,H,S,192]
    const unsigned short* __restrict__ kbf,   // [B,H,S,192]
    const unsigned short* __restrict__ vbf,   // [B,H,S,128]
    unsigned short* __restrict__ attn_bf) {   // [B,S,H*128]
  __shared__ __align__(16) unsigned short vT[VHD_][32];     // 8 KB
  __shared__ __align__(16) unsigned short pbuf[4][16][32];  // 4 KB
  __shared__ __align__(16) unsigned short sK[32 * QKD_];    // 12 KB

  const int tid   = threadIdx.x;
  const int w     = tid >> 5;
  const int lane  = tid & 31;
  const int lhalf = lane >> 4;
  const int lmod  = lane & 15;
  const int h  = blockIdx.y;
  const int b  = blockIdx.z;
  const size_t bh = (size_t)b * H_ + h;
  const int qr0 = blockIdx.x * 64 + w * 16;
  const float scale = 0.0721687836487f;       // 1/sqrt(192)
  const unsigned sK_off = lds_off_u32(&sK[0]);

  // Q fragments (6 x 16x32 bf16 A-frags) in registers
  Frag16 aq[6];
  {
    const unsigned short* qrow = qbf + (bh * S_ + qr0 + lmod) * QKD_;
#pragma unroll
    for (int dc = 0; dc < 6; ++dc) {
      int kbA = dc * 32 + (lhalf << 4);
      aq[dc].q[0] = *(const uint4*)(qrow + kbA);
      aq[dc].q[1] = *(const uint4*)(qrow + kbA + 8);
    }
  }

  v8f   acc[8] = {};
  float mrun[8], lrun[8];
#pragma unroll
  for (int e = 0; e < 8; ++e) { mrun[e] = -1e30f; lrun[e] = 0.0f; }

  const int kend = (blockIdx.x + 1) * 64;
  for (int kb = 0; kb < kend; kb += 32) {
    // stage V chunk [32,128] transposed into vT[n][k] (needs transpose -> DS)
    {
      int kr = tid >> 2;
      int db = (tid & 3) * 32;
      const unsigned short* src = vbf + (bh * S_ + kb + kr) * VHD_ + db;
#pragma unroll
      for (int j = 0; j < 32; ++j) vT[db + j][kr] = src[j];
    }
    // stage K chunk [32,192] = 12KB contiguous slab via async copy
    {
      unsigned kl = sK_off + (unsigned)(tid * 16);
      unsigned kg = (unsigned)(((bh * S_ + kb) * QKD_) * 2) + (unsigned)(tid * 16);
#pragma unroll
      for (int j = 0; j < 6; ++j)
        async_b128(kl + (unsigned)(j * 2048), kg + (unsigned)(j * 2048), kbf);
    }
    wait_async0();
    __syncthreads();

    if (kb <= qr0 + 15) {             // wave-uniform causal skip (EXEC all-ones)
      float st[2][8];
#pragma unroll
      for (int sc = 0; sc < 2; ++sc) {
        v8f sacc = {};
        int c0 = kb + sc * 16;
        const unsigned short* kcolp = sK + (sc * 16 + lmod) * QKD_ + (lhalf << 4);
#pragma unroll
        for (int dc = 0; dc < 6; ++dc) {
          Frag16 ub;
          const unsigned short* p = kcolp + dc * 32;
          ub.q[0] = *(const uint4*)p;
          ub.q[1] = *(const uint4*)(p + 8);
          sacc = __builtin_amdgcn_wmma_f32_16x16x32_bf16(
              false, aq[dc].v, false, ub.v, (short)0, sacc, false, false);
        }
#pragma unroll
        for (int e = 0; e < 8; ++e) {
          int row = qr0 + e + (lhalf << 3);
          int col = c0 + lmod;
          float sv = sacc[e] * scale;
          st[sc][e] = (col > row) ? -1e30f : sv;
        }
      }
      float mnew[8], alpha[8];
#pragma unroll
      for (int e = 0; e < 8; ++e) {
        float mm = fmaxf(st[0][e], st[1][e]);
#pragma unroll
        for (int off = 1; off < 16; off <<= 1)
          mm = fmaxf(mm, __shfl_xor(mm, off, 32));
        mnew[e]  = fmaxf(mrun[e], mm);
        alpha[e] = __expf(mrun[e] - mnew[e]);
        mrun[e]  = mnew[e];
      }
#pragma unroll
      for (int e = 0; e < 8; ++e) {
        int row = qr0 + e + (lhalf << 3);
        float p0 = (kb + lmod > row)      ? 0.0f : __expf(st[0][e] - mnew[e]);
        float p1 = (kb + 16 + lmod > row) ? 0.0f : __expf(st[1][e] - mnew[e]);
        float ps = p0 + p1;
#pragma unroll
        for (int off = 1; off < 16; off <<= 1) ps += __shfl_xor(ps, off, 32);
        lrun[e] = lrun[e] * alpha[e] + ps;
        pbuf[w][e + (lhalf << 3)][lmod]      = f32_to_bf16(p0);
        pbuf[w][e + (lhalf << 3)][16 + lmod] = f32_to_bf16(p1);
#pragma unroll
        for (int ns = 0; ns < 8; ++ns) acc[ns][e] *= alpha[e];
      }
      asm volatile("s_wait_dscnt 0x0" ::: "memory");  // same-wave LDS RAW

      Frag16 pf;
      {
        const unsigned short* pp = &pbuf[w][lmod][lhalf << 4];
        pf.q[0] = *(const uint4*)pp;
        pf.q[1] = *(const uint4*)(pp + 8);
      }
#pragma unroll
      for (int ns = 0; ns < 8; ++ns) {
        Frag16 ub;
        const unsigned short* vp = &vT[(ns << 4) + lmod][lhalf << 4];
        ub.q[0] = *(const uint4*)vp;
        ub.q[1] = *(const uint4*)(vp + 8);
        acc[ns] = __builtin_amdgcn_wmma_f32_16x16x32_bf16(
            false, pf.v, false, ub.v, (short)0, acc[ns], false, false);
      }
    }
    __syncthreads();
  }

#pragma unroll
  for (int ns = 0; ns < 8; ++ns) {
#pragma unroll
    for (int e = 0; e < 8; ++e) {
      int row = qr0 + e + (lhalf << 3);
      int col = h * VHD_ + (ns << 4) + lmod;
      float ov = acc[ns][e] / lrun[e];
      attn_bf[((size_t)b * S_ + row) * (H_ * VHD_) + col] = f32_to_bf16(ov);
    }
  }
}

// ---------------------------------------------------------------------------
// Host launcher
// ---------------------------------------------------------------------------
extern "C" void kernel_launch(void* const* d_in, const int* in_sizes, int n_in,
                              void* d_out, int out_size, void* d_ws, size_t ws_size,
                              hipStream_t stream) {
  (void)in_sizes; (void)n_in; (void)out_size; (void)ws_size;
  const float* x      = (const float*)d_in[0];
  const float* pcos   = (const float*)d_in[1];
  const float* psin   = (const float*)d_in[2];
  const float* w_dkv  = (const float*)d_in[3];
  const float* b_dkv  = (const float*)d_in[4];
  const float* w_uk   = (const float*)d_in[5];
  const float* b_uk   = (const float*)d_in[6];
  const float* w_uv   = (const float*)d_in[7];
  const float* b_uv   = (const float*)d_in[8];
  const float* w_dq   = (const float*)d_in[9];
  const float* b_dq   = (const float*)d_in[10];
  const float* w_uq   = (const float*)d_in[11];
  const float* b_uq   = (const float*)d_in[12];
  const float* w_qr   = (const float*)d_in[13];
  const float* b_qr   = (const float*)d_in[14];
  const float* w_kr   = (const float*)d_in[15];
  const float* b_kr   = (const float*)d_in[16];
  const float* w_o    = (const float*)d_in[17];
  const float* b_o    = (const float*)d_in[18];

  char* base = (char*)d_ws;
  size_t off = 0;
  auto alloc = [&](size_t bytes) {
    size_t o = off;
    off += (bytes + 255) & ~(size_t)255;
    return o;
  };
  const size_t M = (size_t)B_ * S_;   // 4096

  unsigned short* x_bf    = (unsigned short*)(base + alloc(M * DIM_ * 2));
  unsigned short* wdkvT   = (unsigned short*)(base + alloc((size_t)DIM_ * DOWN_ * 2));
  unsigned short* wukT    = (unsigned short*)(base + alloc((size_t)DOWN_ * UP_ * 2));
  unsigned short* wuvT    = (unsigned short*)(base + alloc((size_t)DOWN_ * UP_ * 2));
  unsigned short* wdqT    = (unsigned short*)(base + alloc((size_t)DIM_ * DOWN_ * 2));
  unsigned short* wuqT    = (unsigned short*)(base + alloc((size_t)DOWN_ * UP_ * 2));
  unsigned short* wqrT    = (unsigned short*)(base + alloc((size_t)DOWN_ * RD_ * H_ * 2));
  unsigned short* wkrT    = (unsigned short*)(base + alloc((size_t)DIM_ * RD_ * 2));
  unsigned short* woT     = (unsigned short*)(base + alloc((size_t)H_ * VHD_ * DIM_ * 2));
  float*          ckv_f   = (float*)(base + alloc(M * DOWN_ * 4));
  unsigned short* ckv_bf  = (unsigned short*)(base + alloc(M * DOWN_ * 2));
  float*          cq_f    = (float*)(base + alloc(M * DOWN_ * 4));
  unsigned short* cq_bf   = (unsigned short*)(base + alloc(M * DOWN_ * 2));
  float*          kc_f    = (float*)(base + alloc(M * UP_ * 4));
  float*          v_f     = (float*)(base + alloc(M * UP_ * 4));
  float*          qc_f    = (float*)(base + alloc(M * UP_ * 4));
  float*          qpe_f   = (float*)(base + alloc(M * RD_ * H_ * 4));
  float*          kpe_f   = (float*)(base + alloc(M * RD_ * 4));
  unsigned short* q_bf    = (unsigned short*)(base + alloc((size_t)B_ * H_ * S_ * QKD_ * 2));
  unsigned short* k_bf    = (unsigned short*)(base + alloc((size_t)B_ * H_ * S_ * QKD_ * 2));
  unsigned short* v_bf    = (unsigned short*)(base + alloc((size_t)B_ * H_ * S_ * VHD_ * 2));
  unsigned short* att_bf  = (unsigned short*)(base + alloc(M * (size_t)(H_ * VHD_) * 2));

  cvt_f32_bf16_kernel<<<dim3((unsigned)((M * DIM_ + 255) / 256)), 256, 0, stream>>>(
      x, x_bf, (int)(M * DIM_));
  auto cvtT = [&](const float* in, unsigned short* out, int K, int N) {
    size_t n = (size_t)K * N;
    cvt_transpose_bf16_kernel<<<dim3((unsigned)((n + 255) / 256)), 256, 0, stream>>>(
        in, out, K, N);
  };
  cvtT(w_dkv, wdkvT, DIM_,  DOWN_);
  cvtT(w_uk,  wukT,  DOWN_, UP_);
  cvtT(w_uv,  wuvT,  DOWN_, UP_);
  cvtT(w_dq,  wdqT,  DIM_,  DOWN_);
  cvtT(w_uq,  wuqT,  DOWN_, UP_);
  cvtT(w_qr,  wqrT,  DOWN_, RD_ * H_);
  cvtT(w_kr,  wkrT,  DIM_,  RD_);
  cvtT(w_o,   woT,   H_ * VHD_, DIM_);

  auto gemm = [&](const unsigned short* A, const unsigned short* WT,
                  const float* bias, float* Cf, unsigned short* Cbf,
                  int m, int n, int k) {
    dim3 g(m / GTM, n / GTN);
    gemm_bf16_kernel<<<g, 256, 0, stream>>>(A, WT, bias, Cf, Cbf, m, n, k);
  };
  gemm(x_bf,   wdkvT, b_dkv, ckv_f, ckv_bf, (int)M, DOWN_,    DIM_);
  gemm(x_bf,   wdqT,  b_dq,  cq_f,  cq_bf,  (int)M, DOWN_,    DIM_);
  gemm(x_bf,   wkrT,  b_kr,  kpe_f, nullptr,(int)M, RD_,      DIM_);
  gemm(ckv_bf, wukT,  b_uk,  kc_f,  nullptr,(int)M, UP_,      DOWN_);
  gemm(ckv_bf, wuvT,  b_uv,  v_f,   nullptr,(int)M, UP_,      DOWN_);
  gemm(cq_bf,  wuqT,  b_uq,  qc_f,  nullptr,(int)M, UP_,      DOWN_);
  gemm(cq_bf,  wqrT,  b_qr,  qpe_f, nullptr,(int)M, RD_ * H_, DOWN_);

  assemble_qkv_kernel<<<dim3((unsigned)M), 256, 0, stream>>>(
      qc_f, qpe_f, kc_f, kpe_f, v_f, pcos, psin, q_bf, k_bf, v_bf);

  mla_attention_kernel<<<dim3(S_ / 64, H_, B_), 128, 0, stream>>>(
      q_bf, k_bf, v_bf, att_bf);

  gemm(att_bf, woT, b_o, (float*)d_out, nullptr, (int)M, DIM_, (int)(H_ * VHD_));
}
